// _ReservoirExtractor_6536940224679
// MI455X (gfx1250) — matrix operand order repaired and checked by reading, not compiled, not run on hardware
//
#include <hip/hip_runtime.h>
#include <math.h>

#define B_ 32
#define T_ 512
#define E_ 256
#define R_ 1024
#define L_ 4

typedef unsigned short u16;
typedef unsigned int   u32;
typedef unsigned long long u64;

typedef __attribute__((ext_vector_type(16))) __bf16 v16bf;
typedef __attribute__((ext_vector_type(8)))  float  v8f;
typedef __attribute__((ext_vector_type(4)))  u32    v4u;
typedef __attribute__((ext_vector_type(4)))  int    v4i;
typedef __attribute__((ext_vector_type(8)))  int    v8i;

// ---------------- bf16 helpers (round-to-nearest-even) ----------------
__device__ __forceinline__ u16 f2bf(float f) {
  u32 u = __builtin_bit_cast(u32, f);
  u += 0x7FFFu + ((u >> 16) & 1u);
  return (u16)(u >> 16);
}
__device__ __forceinline__ float bf2f(u16 b) {
  return __builtin_bit_cast(float, (u32)b << 16);
}

// ---------------- setup: pack transposed recurrent weights ----------------
// Packed layout per (layer,tile) chunk of 32768 bf16 (64KB):
//   [((j*2 + n16)*32 + lane)*16 + e]  where
//   n = tile*32 + n16*16 + (lane&15)   (output column)
//   k = j*32 + (lane>>4)*16 + e        (input index)
// This is the contiguous-k per-lane B operand order (32x16 bf16) this code uses, matching load_a,
// so the hot loop reads contiguous 32B per lane from LDS.
__global__ void pack_rec_kernel(const float* __restrict__ Rst, u16* __restrict__ prec) {
  u64 i = (u64)blockIdx.x * blockDim.x + threadIdx.x;   // 0 .. 4*32*32768-1
  u32 e    = (u32)i & 15u;
  u32 r    = (u32)(i >> 4);
  u32 lane = r & 31u; r >>= 5;
  u32 n16  = r & 1u;  r >>= 1;
  u32 j    = r & 31u; r >>= 5;
  u32 tile = r & 31u; r >>= 5;
  u32 l    = r;
  u32 n = (tile << 5) + (n16 << 4) + (lane & 15u);
  u32 k = (j << 5) + ((lane >> 4) << 4) + e;
  prec[i] = f2bf(Rst[((u64)l * R_ + k) * R_ + n]);
}

// Input weights: layer 0 uses Win0 [E,R] (only j<8), layers 1..3 use Win_rest.
__global__ void pack_in_kernel(const float* __restrict__ Win0,
                               const float* __restrict__ WinR,
                               u16* __restrict__ pin) {
  u64 i = (u64)blockIdx.x * blockDim.x + threadIdx.x;
  u32 e    = (u32)i & 15u;
  u32 r    = (u32)(i >> 4);
  u32 lane = r & 31u; r >>= 5;
  u32 n16  = r & 1u;  r >>= 1;
  u32 j    = r & 31u; r >>= 5;
  u32 tile = r & 31u; r >>= 5;
  u32 l    = r;
  if (l == 0 && j >= 8u) return;               // layer0: K=256 only
  u32 n = (tile << 5) + (n16 << 4) + (lane & 15u);
  u32 k = (j << 5) + ((lane >> 4) << 4) + e;
  float v = (l == 0) ? Win0[(u64)k * R_ + n]
                     : WinR[(((u64)(l - 1)) * R_ + k) * R_ + n];
  pin[i] = f2bf(v);
}

// Gather embeddings -> xe[t][b][e] bf16
__global__ void pack_xe_kernel(const int* __restrict__ x,
                               const float* __restrict__ embed,
                               u16* __restrict__ xe) {
  u64 i = (u64)blockIdx.x * blockDim.x + threadIdx.x;  // T*B*E
  u32 e = (u32)i & 255u;
  u32 b = ((u32)(i >> 8)) & 31u;
  u32 t = (u32)(i >> 13);
  int tok = x[(u64)b * T_ + t];
  xe[i] = f2bf(embed[(u64)tok * E_ + e]);
}

// ---------------- TDM: load a linear chunk into LDS (data_size = 8B) ----------------
__device__ __forceinline__ void tdm_load(u32 lds_off, const void* gptr, u32 nelem8) {
  u64 ga = (u64)(size_t)gptr;
  v4u g0;
  g0.x = 1u;                                         // count=1, user descriptor
  g0.y = lds_off;                                    // LDS byte address
  g0.z = (u32)ga;                                    // global_addr[31:0]
  g0.w = (u32)((ga >> 32) & 0x1FFFFFFu) | (2u << 30); // global_addr[56:32] | type=2
  v8i g1;
  g1[0] = (int)(3u << 16);                           // data_size=3 (8B), mask=0, no pad
  g1[1] = (int)((nelem8 & 0xFFFFu) << 16);           // tensor_dim0[15:0]
  g1[2] = (int)((nelem8 >> 16) | (1u << 16));        // tensor_dim0[31:16] | tensor_dim1=1
  g1[3] = (int)((nelem8 & 0xFFFFu) << 16);           // tile_dim0 = nelem8
  g1[4] = 0;                                         // tile_dim1=0, tile_dim2=0
  g1[5] = (int)nelem8;                               // tensor_dim0_stride lo
  g1[6] = 0;
  g1[7] = 0;
  v4i gz = {0, 0, 0, 0};
#if defined(__clang_major__) && (__clang_major__ >= 23)
  v8i gz8 = {0, 0, 0, 0, 0, 0, 0, 0};
  __builtin_amdgcn_tensor_load_to_lds(g0, g1, gz, gz, gz8, 0);
#else
  __builtin_amdgcn_tensor_load_to_lds(g0, g1, gz, gz, 0);
#endif
}

// ---------------- A-operand load: 16x32 bf16 tile in WMMA lane order ----------------
// lane<16: elems 0..15 = K+0..15 (row = m16+lane)
// lane>=16: elems 0..15 = K+16..31 (caller pre-adds +16 via asel)
__device__ __forceinline__ v16bf load_a(const u16* p) {
  union { uint4 q[2]; v16bf v; } u_;
  u_.q[0] = *(const uint4*)(p);
  u_.q[1] = *(const uint4*)(p + 8);
  return u_.v;
}

__device__ __forceinline__ void wait_done(int* p) {
  while (__hip_atomic_load(p, __ATOMIC_RELAXED, __HIP_MEMORY_SCOPE_AGENT) < 128) {
    __builtin_amdgcn_s_sleep(1);
  }
  (void)__hip_atomic_load(p, __ATOMIC_ACQUIRE, __HIP_MEMORY_SCOPE_AGENT);
}

// ---------------- persistent pipelined ESN kernel ----------------
// grid = 128 blocks: block = (layer l = bx>>5, column tile = bx&31), 128 threads (4 waves).
// Wave w computes the 16x16 output tile (m16 = (w&1)*16, n16 = (w>>1)*16).
// Weight slices live in LDS (rec 64KB @0, input 64KB @65536) for all 512 steps.
__global__ void __launch_bounds__(128) esn_kernel(
    const u16* __restrict__ prec, const u16* __restrict__ pin,
    const u16* __restrict__ xe, u16* states, int* done) {
  extern __shared__ char smem[];
  const u32 l    = blockIdx.x >> 5;
  const u32 tile = blockIdx.x & 31u;
  const u32 tid  = threadIdx.x;
  const u32 wave = tid >> 5;
  const u32 lane = tid & 31u;
  const u32 m16  = (wave & 1u) << 4;
  const u32 n16i = wave >> 1;
  const u32 nbase = tile << 5;
  const u32 in_jmax = (l == 0) ? 8u : 32u;

  if (wave == 0) {  // TDM: one wave issues both DMA copies, EXEC-independent
    tdm_load((u32)(size_t)(void*)smem, prec + (u64)blockIdx.x * 32768u, 8192u);
    tdm_load((u32)(size_t)(void*)smem + 65536u, pin + (u64)blockIdx.x * 32768u,
             (l == 0) ? 2048u : 8192u);
    __builtin_amdgcn_s_wait_tensorcnt(0);
  }
  __syncthreads();

  const u32 arow = m16 + (lane & 15u);
  const u32 asel = (lane >> 4) << 4;                  // 0 or 16 (K phase per half-wave)
  const u32 boff = (n16i << 10) + (lane << 5);        // bytes within a 2KB k-step chunk
  u16* st_l = states + ((u64)l * 4u) * (B_ * R_);

  for (int t = 0; t < T_; ++t) {
    const u32 pprev = (u32)(t + 3) & 3u;
    const u32 pcur  = (u32)t & 3u;
    if (t >= 1)                 wait_done(&done[l * T_ + (t - 1)]);       // own s(t-1) full
    if (l > 0)                  wait_done(&done[(l - 1) * T_ + t]);       // input ready
    if (l < L_ - 1 && t >= 4)   wait_done(&done[(l + 1) * T_ + (t - 4)]); // buffer reuse

    v8f acc = {0.f, 0.f, 0.f, 0.f, 0.f, 0.f, 0.f, 0.f};

    // recurrent GEMM: s_l(t-1)[32,1024] x R_l[:, tile]
    const u16* aptr = st_l + (u64)pprev * (B_ * R_) + (u64)arow * R_ + asel;
    for (u32 j = 0; j < 32u; ++j) {
      v16bf a = load_a(aptr + j * 32u);
      v16bf b = *(const v16bf*)(smem + (j << 11) + boff);
      acc = __builtin_amdgcn_wmma_f32_16x16x32_bf16(false, a, false, b,
                                                    (short)0, acc, false, false);
    }
    // input GEMM: (xe_t or s_{l-1}(t)) x W_l[:, tile]
    const u16* iptr = (l == 0)
        ? (xe + ((u64)t * B_ + arow) * E_ + asel)
        : (states + (((u64)(l - 1) * 4u + pcur) * B_ + arow) * R_ + asel);
    const char* smem_in = smem + 65536;
    for (u32 j = 0; j < in_jmax; ++j) {
      v16bf a = load_a(iptr + j * 32u);
      v16bf b = *(const v16bf*)(smem_in + (j << 11) + boff);
      acc = __builtin_amdgcn_wmma_f32_16x16x32_bf16(false, a, false, b,
                                                    (short)0, acc, false, false);
    }
    // tanh + store s_l(t) (D layout: VGPR r -> M = r + 8*(lane>=16), N = lane&15)
    u16* sout = st_l + (u64)pcur * (B_ * R_);
    const u32 orow = m16 + ((lane >> 4) << 3);
    const u32 ocol = nbase + (n16i << 4) + (lane & 15u);
#pragma unroll
    for (int r = 0; r < 8; ++r) {
      sout[(u64)(orow + r) * R_ + ocol] = f2bf(tanhf(acc[r]));
    }
    if (lane == 0) {  // per-wave release: orders this wave's stores device-wide
      __hip_atomic_fetch_add(&done[l * T_ + t], 1, __ATOMIC_RELEASE,
                             __HIP_MEMORY_SCOPE_AGENT);
    }
  }
}

// ---------------- LayerNorm over concat[s0..s3](t=511) -> fp32 out ----------------
__global__ void ln_kernel(const u16* __restrict__ states,
                          const float* __restrict__ gamma,
                          const float* __restrict__ beta,
                          float* __restrict__ out) {
  __shared__ float red[256];
  const u32 b = blockIdx.x, tid = threadIdx.x;
  float loc[16];
  float s = 0.f;
#pragma unroll
  for (int i = 0; i < 16; ++i) {
    u32 idx = (u32)i * 256u + tid;            // 0..4095
    u32 l = idx >> 10, r = idx & 1023u;
    float v = bf2f(states[(((u64)l * 4u + 3u) * B_ + b) * R_ + r]); // parity 511%4 == 3
    loc[i] = v;
    s += v;
  }
  red[tid] = s; __syncthreads();
  for (int o = 128; o > 0; o >>= 1) { if (tid < (u32)o) red[tid] += red[tid + o]; __syncthreads(); }
  const float mu = red[0] * (1.f / 4096.f);
  __syncthreads();
  float vs = 0.f;
#pragma unroll
  for (int i = 0; i < 16; ++i) { float d = loc[i] - mu; vs += d * d; }
  red[tid] = vs; __syncthreads();
  for (int o = 128; o > 0; o >>= 1) { if (tid < (u32)o) red[tid] += red[tid + o]; __syncthreads(); }
  const float inv = rsqrtf(red[0] * (1.f / 4096.f) + 1e-5f);
#pragma unroll
  for (int i = 0; i < 16; ++i) {
    u32 idx = (u32)i * 256u + tid;
    out[(u64)b * 4096u + idx] = (loc[i] - mu) * inv * gamma[idx] + beta[idx];
  }
}

// ---------------- host ----------------
extern "C" void kernel_launch(void* const* d_in, const int* in_sizes, int n_in,
                              void* d_out, int out_size, void* d_ws, size_t ws_size,
                              hipStream_t stream) {
  (void)in_sizes; (void)n_in; (void)out_size; (void)ws_size;
  const int*   x        = (const int*)d_in[0];
  const float* embed    = (const float*)d_in[1];
  const float* Win0     = (const float*)d_in[2];
  const float* Win_rest = (const float*)d_in[3];
  const float* R_stack  = (const float*)d_in[4];
  const float* gamma    = (const float*)d_in[5];
  const float* beta     = (const float*)d_in[6];

  char* ws = (char*)d_ws;
  u16* prec   = (u16*)(ws);                         // 8 MB packed recurrent weights
  u16* pin    = (u16*)(ws + (8ull << 20));          // 8 MB packed input weights
  u16* xe     = (u16*)(ws + (16ull << 20));         // 8 MB bf16 embeddings [T][B][E]
  u16* states = (u16*)(ws + (24ull << 20));         // 1 MB states [L][4][B][R]
  int* done   = (int*)(ws + (25ull << 20));         // 8 KB counters [L][T]

  // zero states (quad-buffer parity 3 = s(-1) = 0) and done counters (adjacent)
  hipMemsetAsync(states, 0, (1ull << 20) + (size_t)L_ * T_ * sizeof(int), stream);

  const int PACK_N = L_ * 32 * 32768 / 256;         // 16384 blocks
  pack_rec_kernel<<<PACK_N, 256, 0, stream>>>(R_stack, prec);
  pack_in_kernel<<<PACK_N, 256, 0, stream>>>(Win0, Win_rest, pin);
  pack_xe_kernel<<<T_ * B_ * E_ / 256, 256, 0, stream>>>(x, embed, xe);

  hipFuncSetAttribute(reinterpret_cast<const void*>(esn_kernel),
                      hipFuncAttributeMaxDynamicSharedMemorySize, 131072);
  esn_kernel<<<L_ * 32, 128, 131072, stream>>>(prec, pin, xe, states, done);

  ln_kernel<<<B_, 256, 0, stream>>>(states, gamma, beta, (float*)d_out);
}
